// SelfAttention_49022756717127
// MI455X (gfx1250) — compile-verified, matrix operand order checked by reading, not run on hardware
//
#include <hip/hip_runtime.h>

// ---------------------------------------------------------------------------
// MI455X (gfx1250) self-attention pipeline, wave32 + v_wmma_f32_16x16x32_f16.
// Round 4: TDM (tensor_load_to_lds, 6-arg clang-23 form) stages the attention
// K-chunk into LDS with hardware padding (pitch 72 halves), synchronized via
// s_wait_tensorcnt. DPP softmax reductions, 32x64-per-wave GEMM blocking.
// ---------------------------------------------------------------------------

typedef __attribute__((ext_vector_type(16))) _Float16 v16h;
typedef __attribute__((ext_vector_type(8)))  _Float16 v8h;
typedef __attribute__((ext_vector_type(8)))  float    v8f;
typedef __attribute__((ext_vector_type(4)))  unsigned int v4u;
typedef __attribute__((ext_vector_type(8)))  int      v8i;
typedef __attribute__((ext_vector_type(4)))  int      v4i;

static constexpr int BB  = 2;
static constexpr int TT  = 2048;
static constexpr int DD  = 1024;
static constexpr int HH  = 16;
static constexpr int HDD = 64;
static constexpr int REG = 16;
static constexpr int HALFWIN = 128;   // WIN/2

__device__ __forceinline__ v8f wmma16(v16h a, v16h b, v8f c) {
  return __builtin_amdgcn_wmma_f32_16x16x32_f16(false, a, false, b, (short)0, c,
                                                false, false);
}

// ---- DPP lane-xor helpers (reduction across the 16 lanes holding one row) --
static constexpr int kSelXor1 =
    (1 << 0) | (0 << 3) | (3 << 6) | (2 << 9) | (5 << 12) | (4 << 15) | (7 << 18) | (6 << 21);
static constexpr int kSelXor2 =
    (2 << 0) | (3 << 3) | (0 << 6) | (1 << 9) | (6 << 12) | (7 << 15) | (4 << 18) | (5 << 21);
static constexpr int kSelXor4 =
    (4 << 0) | (5 << 3) | (6 << 6) | (7 << 9) | (0 << 12) | (1 << 15) | (2 << 18) | (3 << 21);

template <int SEL>
__device__ __forceinline__ float dpp8f(float x) {
  union { float f; int i; } u, r;
  u.f = x;
  r.i = __builtin_amdgcn_mov_dpp8(u.i, SEL);
  return r.f;
}
// xor8 within a 16-lane row == DPP16 row_ror:8 (0x128)
__device__ __forceinline__ float rowxor8f(float x) {
  union { float f; int i; } u, r;
  u.f = x;
  r.i = __builtin_amdgcn_update_dpp(0, u.i, 0x128, 0xf, 0xf, true);
  return r.f;
}
__device__ __forceinline__ float redmax16(float v) {
  v = fmaxf(v, dpp8f<kSelXor1>(v));
  v = fmaxf(v, dpp8f<kSelXor2>(v));
  v = fmaxf(v, dpp8f<kSelXor4>(v));
  v = fmaxf(v, rowxor8f(v));
  return v;
}
__device__ __forceinline__ float redsum16(float v) {
  v += dpp8f<kSelXor1>(v);
  v += dpp8f<kSelXor2>(v);
  v += dpp8f<kSelXor4>(v);
  v += rowxor8f(v);
  return v;
}

// Load a 16x32 f16 fragment (A-layout; also B when tile is N-major,
// K-contiguous). Per CDNA5 ISA 7.12.2:
//   lanes 0-15 : row = lane,    K = {0..7, 16..23}
//   lanes 16-31: row = lane-16, K = {8..15, 24..31}
__device__ __forceinline__ v16h load_frag16(const _Float16* tile, int pitch) {
  const int lane = threadIdx.x & 31;
  const int row  = lane & 15;
  const int kh   = (lane >> 4) << 3;
  const _Float16* p = tile + row * pitch;
  union { v16h v; v8h h[2]; } u;
  u.h[0] = *(const v8h*)(p + kh);
  u.h[1] = *(const v8h*)(p + 16 + kh);
  return u.v;
}

// ---- TDM: load a rows x 64-half tile (row-major, stride 64) into LDS with
// hardware padding of 4 DWORDs after every 32 DWORDs -> LDS pitch 72 halves.
// D# layout per CDNA5 ISA 8.3/8.4. clang-23 6-arg builtin form.
__device__ __forceinline__ void tdm_load_rows64(const _Float16* gsrc,
                                                unsigned lds_byte_addr,
                                                int rows, int rows_avail) {
  const unsigned long long ga = (unsigned long long)(size_t)gsrc;
  const v4u g0 = {
      1u,                                          // count=1, user descriptor
      lds_byte_addr,                               // lds_addr
      (unsigned)(ga & 0xFFFFFFFFu),                // global_addr[31:0]
      (unsigned)((ga >> 32) & 0x01FFFFFFu) | (2u << 30)  // addr[56:32] | type=2
  };
  const v8i g1 = {
      (1 << 16) | (1 << 20) | (4 << 22) | (3 << 25),
      //  data_size=2B | pad_enable | pad_interval=32DW | pad_amount=4DW
      (int)(64u << 16),                            // tensor_dim0 = 64
      (int)((unsigned)rows_avail << 16),           // tensor_dim1 (lo16)
      (int)(64u << 16),                            // tile_dim0 = 64
      rows,                                        // tile_dim1 = rows
      64,                                          // tensor_dim0_stride = 64
      0, 0
  };
  const v4i gz4 = {0, 0, 0, 0};
  const v8i gz8 = {0, 0, 0, 0, 0, 0, 0, 0};
  __builtin_amdgcn_tensor_load_to_lds(g0, g1, gz4, gz4, gz8, 0);
}

// ---------------------------------------------------------------------------
// GEMM: C[m][n] = sum_k A[m][k] * W[n][k]   (W row-major [N][K], fp32)
// Block tile 128(M) x 128(N), K-step 32. 8 waves = 4(M) x 2(N); each wave
// computes 32x64 -> 8 WMMA per K-step from 6 fragment loads.
// ---------------------------------------------------------------------------
template <typename AT, typename OT>
__global__ __launch_bounds__(256)
void gemm_xwT_kernel(const AT* __restrict__ A, const float* __restrict__ W,
                     OT* __restrict__ C, int M, int N, int K) {
  __shared__ __align__(16) _Float16 As[128 * 48];  // pitch 48 halves (96B)
  __shared__ __align__(16) _Float16 Bs[128 * 48];  // transposed: Bs[n][k]

  const int tid  = threadIdx.x;
  const int wave = tid >> 5;
  const int lane = tid & 31;
  const int mw = wave >> 1;              // 0..3
  const int nw = wave & 1;               // 0..1
  const int m0 = blockIdx.y * 128;
  const int n0 = blockIdx.x * 128;

  v8f acc[2][4] = {};

  for (int k0 = 0; k0 < K; k0 += 32) {
#pragma unroll
    for (int c = 0; c < 2; ++c) {
      const int cid = tid * 2 + c;       // 0..511
      const int r   = cid >> 2;          // 0..127
      const int ko  = (cid & 3) << 3;    // 0,8,16,24
      const AT*    srcA = A + (size_t)(m0 + r) * K + k0 + ko;
      const float* srcB = W + (size_t)(n0 + r) * K + k0 + ko;
      __builtin_prefetch(srcA + 32, 0, 0);
      __builtin_prefetch(srcB + 32, 0, 0);
      _Float16* dA = &As[r * 48 + ko];
      _Float16* dB = &Bs[r * 48 + ko];
#pragma unroll
      for (int u = 0; u < 8; ++u) {
        dA[u] = (_Float16)srcA[u];
        dB[u] = (_Float16)srcB[u];
      }
    }
    __syncthreads();

    const v16h a0 = load_frag16(&As[(mw * 32) * 48],      48);
    const v16h a1 = load_frag16(&As[(mw * 32 + 16) * 48], 48);
#pragma unroll
    for (int nt = 0; nt < 4; ++nt) {
      const v16h b = load_frag16(&Bs[(nw * 64 + nt * 16) * 48], 48);
      acc[0][nt] = wmma16(a0, b, acc[0][nt]);
      acc[1][nt] = wmma16(a1, b, acc[1][nt]);
    }
    __syncthreads();
  }

  const int hl = lane >> 4, nlo = lane & 15;
#pragma unroll
  for (int mt = 0; mt < 2; ++mt)
#pragma unroll
    for (int nt = 0; nt < 4; ++nt)
#pragma unroll
      for (int r = 0; r < 8; ++r) {
        const int m = m0 + mw * 32 + mt * 16 + r + 8 * hl;
        const int n = n0 + nw * 64 + nt * 16 + nlo;
        C[(size_t)m * N + n] = (OT)acc[mt][nt][r];
      }
}

// ---------------------------------------------------------------------------
// RoPE + head split: qkv[B*T][3D] f16 -> Q,K (roped), V : [B,H,T,64] f16
// ---------------------------------------------------------------------------
__global__ __launch_bounds__(256)
void rope_split_kernel(const _Float16* __restrict__ qkv,
                       _Float16* __restrict__ Q,
                       _Float16* __restrict__ Kd,
                       _Float16* __restrict__ V) {
  const int idx = blockIdx.x * 256 + threadIdx.x;
  const int p = idx & 31;
  const int h = (idx >> 5) & (HH - 1);
  const int t = (idx >> 9) & (TT - 1);
  const int b = idx >> 20;
  if (b >= BB) return;

  const _Float16* row = qkv + (size_t)(b * TT + t) * (3 * DD);
  const int e = h * HDD + 2 * p;

  const float theta = __expf(-(float)(2 * p) * (9.210340371976184f / 64.0f));
  const float ang = (float)t * theta;
  float s, c;
  __sincosf(ang, &s, &c);

  const size_t o = ((size_t)(b * HH + h) * TT + t) * HDD + 2 * p;

  const float q0 = (float)row[e], q1 = (float)row[e + 1];
  Q[o]     = (_Float16)(q0 * c - q1 * s);
  Q[o + 1] = (_Float16)(q1 * c + q0 * s);

  const float k0 = (float)row[DD + e], k1 = (float)row[DD + e + 1];
  Kd[o]     = (_Float16)(k0 * c - k1 * s);
  Kd[o + 1] = (_Float16)(k1 * c + k0 * s);

  V[o]     = row[2 * DD + e];
  V[o + 1] = row[2 * DD + e + 1];
}

// ---------------------------------------------------------------------------
// Masked flash attention. Block = (b,h,128 queries), 8 waves x 16-row q-tiles,
// streaming 64-key chunks. K-chunk staged by the Tensor Data Mover (wave 0,
// TENSORcnt), V transposed manually. 16 WMMA per chunk per wave.
// keep(i,j) = !( j<16 | i<16 | (j in [max(16,i-128), i+128]) )
// ---------------------------------------------------------------------------
__global__ __launch_bounds__(256)
void attn_kernel(const _Float16* __restrict__ Q,
                 const _Float16* __restrict__ Kd,
                 const _Float16* __restrict__ V,
                 _Float16* __restrict__ Y) {
  __shared__ __align__(16) _Float16 Ks[64 * 72];     // K chunk (TDM, pitch 72)
  __shared__ __align__(16) _Float16 VT[64 * 72];     // V chunk transposed [hd][key]
  __shared__ __align__(16) _Float16 Ps[8][16 * 72];  // wave-private P staging

  const int tid  = threadIdx.x;
  const int wave = tid >> 5;
  const int lane = tid & 31;
  const int hl   = lane >> 4;
  const int nlo  = lane & 15;

  const int qtiles = TT / 128;                // 16
  const int bh = blockIdx.x / qtiles;
  const int qb = (blockIdx.x % qtiles) * 128;
  const int b  = bh / HH;
  const int h  = bh % HH;

  const _Float16* Qp = Q  + (size_t)bh * TT * HDD;
  const _Float16* Kp = Kd + (size_t)bh * TT * HDD;
  const _Float16* Vp = V  + (size_t)bh * TT * HDD;

  const int i0 = qb + wave * 16;
  v16h qf0 = load_frag16(Qp + (size_t)i0 * HDD,      HDD);
  v16h qf1 = load_frag16(Qp + (size_t)i0 * HDD + 32, HDD);
#pragma unroll
  for (int u = 0; u < 16; ++u) {             // fold 1/sqrt(64) into Q (exact)
    qf0[u] = qf0[u] * (_Float16)0.125f;
    qf1[u] = qf1[u] * (_Float16)0.125f;
  }

  // per-row mask bounds, hoisted out of the chunk loop
  int lo_r[8], hi_r[8];
  bool iblk[8];
#pragma unroll
  for (int r = 0; r < 8; ++r) {
    const int i = i0 + r + 8 * hl;
    lo_r[r] = (i - HALFWIN) > REG ? (i - HALFWIN) : REG;
    hi_r[r] = i + HALFWIN;
    iblk[r] = (i < REG);
  }

  v8f o[4] = {};
  float mrow[8], lrow[8];
#pragma unroll
  for (int r = 0; r < 8; ++r) { mrow[r] = -1e30f; lrow[r] = 0.0f; }

  const unsigned ksLds = (unsigned)(size_t)&Ks[0];

  for (int j0 = 0; j0 < TT; j0 += 64) {
    // ---- K chunk via Tensor Data Mover (one wave issues; EXEC-independent) -
    if (wave == 0) {
      tdm_load_rows64(Kp + (size_t)j0 * HDD, ksLds, 64, TT - j0);
      __builtin_amdgcn_s_wait_tensorcnt(0);
    }
    // ---- V chunk: manual transpose into VT -------------------------------
    {
      const int kr = tid >> 2;              // 0..63 key in chunk
      const int ko = (tid & 3) << 4;        // 0,16,32,48
      const _Float16* vsrc = Vp + (size_t)(j0 + kr) * HDD + ko;
      const v8h v0 = *(const v8h*)vsrc;
      const v8h v1 = *(const v8h*)(vsrc + 8);
#pragma unroll
      for (int u = 0; u < 8; ++u) {
        VT[(ko + u) * 72 + kr]     = v0[u];
        VT[(ko + 8 + u) * 72 + kr] = v1[u];
      }
    }
    __syncthreads();

    // whole-chunk skip: entire 16x64 tile is blocked
    const bool skip =
        (i0 + 15 < REG) ||
        (j0 >= REG && j0 >= i0 + 15 - HALFWIN && (j0 + 63) <= i0 + HALFWIN);

    if (!skip) {
      float pe[4][8];
      float rmax[8];
#pragma unroll
      for (int r = 0; r < 8; ++r) rmax[r] = -1e30f;

#pragma unroll
      for (int st = 0; st < 4; ++st) {
        v8f s = {};
        s = wmma16(qf0, load_frag16(&Ks[st * 16 * 72],      72), s);
        s = wmma16(qf1, load_frag16(&Ks[st * 16 * 72 + 32], 72), s);
        const int j = j0 + st * 16 + nlo;
        const bool jlt = (j < REG);
#pragma unroll
        for (int r = 0; r < 8; ++r) {
          const bool blocked = jlt | iblk[r] | ((j >= lo_r[r]) & (j <= hi_r[r]));
          const float sv = blocked ? -1e30f : s[r];
          pe[st][r] = sv;
          rmax[r]   = fmaxf(rmax[r], sv);
        }
      }
#pragma unroll
      for (int r = 0; r < 8; ++r) rmax[r] = redmax16(rmax[r]);

#pragma unroll
      for (int r = 0; r < 8; ++r) {
        const float mn = fmaxf(mrow[r], rmax[r]);
        const float al = __expf(mrow[r] - mn);
        mrow[r] = mn;
        float ps = 0.0f;
#pragma unroll
        for (int st = 0; st < 4; ++st) {
          const float sv = pe[st][r];
          const float pv = (sv > -1e29f) ? __expf(sv - mn) : 0.0f;
          pe[st][r] = pv;
          ps += pv;
        }
        ps = redsum16(ps);
        lrow[r] = lrow[r] * al + ps;
#pragma unroll
        for (int nt = 0; nt < 4; ++nt) o[nt][r] *= al;
      }

      // stage P (C-layout -> A-layout) through wave-private LDS
#pragma unroll
      for (int st = 0; st < 4; ++st)
#pragma unroll
        for (int r = 0; r < 8; ++r)
          Ps[wave][(r + 8 * hl) * 72 + st * 16 + nlo] = (_Float16)pe[st][r];

      const v16h pf0 = load_frag16(&Ps[wave][0],  72);   // keys 0..31
      const v16h pf1 = load_frag16(&Ps[wave][32], 72);   // keys 32..63
#pragma unroll
      for (int nt = 0; nt < 4; ++nt) {
        o[nt] = wmma16(pf0, load_frag16(&VT[nt * 16 * 72],      72), o[nt]);
        o[nt] = wmma16(pf1, load_frag16(&VT[nt * 16 * 72 + 32], 72), o[nt]);
      }
    }
    __syncthreads();
  }

  // finalize: y[b][t][h*64 + hd] = o / max(l, 1e-20), f16
#pragma unroll
  for (int r = 0; r < 8; ++r) {
    const float inv = 1.0f / fmaxf(lrow[r], 1e-20f);
    const int tg = i0 + r + 8 * hl;
    const size_t base = ((size_t)b * TT + tg) * DD + h * HDD;
#pragma unroll
    for (int nt = 0; nt < 4; ++nt)
      Y[base + nt * 16 + nlo] = (_Float16)(o[nt][r] * inv);
  }
}

// ---------------------------------------------------------------------------
// Host-side orchestration (graph-capture safe: only kernel launches on stream)
// ---------------------------------------------------------------------------
extern "C" void kernel_launch(void* const* d_in, const int* in_sizes, int n_in,
                              void* d_out, int out_size, void* d_ws, size_t ws_size,
                              hipStream_t stream) {
  (void)in_sizes; (void)n_in; (void)out_size; (void)ws_size;

  const float* x      = (const float*)d_in[0];   // [B,T,D]
  const float* w_attn = (const float*)d_in[1];   // [3D,D]
  const float* w_proj = (const float*)d_in[2];   // [D,D]
  float* out = (float*)d_out;                    // [B,T,D]

  char* ws = (char*)d_ws;
  const size_t M = (size_t)BB * TT;              // 4096
  _Float16* qkv = (_Float16*)ws;                        // M x 3D  (24 MB)
  _Float16* Q   = (_Float16*)(ws + M * 3 * DD * sizeof(_Float16));
  _Float16* K   = Q + (size_t)BB * HH * TT * HDD;       // 8 MB each
  _Float16* V   = K + (size_t)BB * HH * TT * HDD;
  _Float16* Y   = V + (size_t)BB * HH * TT * HDD;       // M x D (8 MB)

  // 1) QKV = x @ w_attn^T : 4096 x 3072 x 1024
  dim3 g1(3 * DD / 128, M / 128);
  gemm_xwT_kernel<float, _Float16><<<g1, 256, 0, stream>>>(
      x, w_attn, qkv, (int)M, 3 * DD, DD);

  // 2) RoPE + split into Q/K/V [B,H,T,64]
  const int nrope = BB * TT * HH * (HDD / 2);
  rope_split_kernel<<<nrope / 256, 256, 0, stream>>>(qkv, Q, K, V);

  // 3) masked flash attention -> Y [B,T,D] f16
  attn_kernel<<<BB * HH * (TT / 128), 256, 0, stream>>>(Q, K, V, Y);

  // 4) out = Y @ w_proj^T : 4096 x 1024 x 1024, fp32 out
  dim3 g4(DD / 128, M / 128);
  gemm_xwT_kernel<_Float16, float><<<g4, 256, 0, stream>>>(
      Y, w_proj, out, (int)M, DD, DD);
}
